// MultiLayerGAT_9895604650471
// MI455X (gfx1250) — compile-verified, matrix operand order checked by reading, not run on hardware
//
#include <hip/hip_runtime.h>
#include <hip/hip_bf16.h>
#include <math.h>

typedef _Float16 f16;
typedef __attribute__((ext_vector_type(16))) _Float16 v16h;
typedef __attribute__((ext_vector_type(8)))  float    v8f;

#define GN     10000      // nodes
#define GE     320000     // raw edges
#define GET    330000     // edges + self loops
#define GIN    128
#define GH     8
#define GC     32
#define GHC    256
#define GOUT   40
#define GOUTP  48
#define GNPAD  10016      // 626 * 16

// ---------------- float <-> monotone uint encoding for atomic float max ----
__device__ __forceinline__ unsigned enc_f(float f) {
  unsigned u = __float_as_uint(f);
  return (u >> 31) ? ~u : (u | 0x80000000u);
}
__device__ __forceinline__ float dec_f(unsigned u) {
  return __uint_as_float((u >> 31) ? (u & 0x7FFFFFFFu) : ~u);
}

// ---------------- elementwise helpers --------------------------------------
__global__ void fill_f32(float* __restrict__ p, int n, float v) {
  int i = blockIdx.x * blockDim.x + threadIdx.x;
  if (i < n) p[i] = v;
}

// f32 [rows][cols] -> f16 [rowsPad][cols], zero padding rows
__global__ void cvt_pad_f16(const float* __restrict__ src, f16* __restrict__ dst,
                            int rows, int rowsPad, int cols) {
  int i = blockIdx.x * blockDim.x + threadIdx.x;
  if (i >= rowsPad * cols) return;
  int r = i / cols;
  dst[i] = (r < rows) ? (f16)src[i] : (f16)0.0f;
}

// W f32 [K][Ncols] -> Wt f16 [NcolPad][K] (row n of Wt = column n of W)
__global__ void transpose_cvt_w(const float* __restrict__ W, f16* __restrict__ Wt,
                                int K, int Ncols, int NcolPad) {
  int i = blockIdx.x * blockDim.x + threadIdx.x;
  if (i >= NcolPad * K) return;
  int n = i / K, k = i - n * K;
  Wt[i] = (n < Ncols) ? (f16)W[(size_t)k * Ncols + n] : (f16)0.0f;
}

// ---------------- WMMA GEMM: C[M][ldc] = A[Mpad][K] * Bt[NcolPad][K]^T ------
union Frag { v16h v; uint4 q[2]; };

__global__ void gat_wmma_gemm(const f16* __restrict__ A, const f16* __restrict__ Bt,
                              float* __restrict__ C, int M, int Ncols, int ldc,
                              int K, int Mtiles, int Ntiles) {
  int wave = (int)((blockIdx.x * blockDim.x + threadIdx.x) >> 5);
  int lane = threadIdx.x & 31;
  if (wave >= Mtiles * Ntiles) return;          // wave-uniform: EXEC all-1 inside
  int tm = wave / Ntiles, tn = wave - tm * Ntiles;
  int r = lane & 15, g = lane >> 4;             // half-wave group selects K sub-block

  const f16* Ap = A  + (size_t)(tm * 16 + r) * K + g * 8;
  const f16* Bp = Bt + (size_t)(tn * 16 + r) * K + g * 8;

  v8f acc = {};
  for (int k0 = 0; k0 < K; k0 += 32) {
    Frag a, b;
    a.q[0] = *(const uint4*)(Ap + k0);          // K = k0 + g*8 + 0..7
    a.q[1] = *(const uint4*)(Ap + k0 + 16);     // K = k0 + 16 + g*8 + 0..7
    b.q[0] = *(const uint4*)(Bp + k0);
    b.q[1] = *(const uint4*)(Bp + k0 + 16);
    acc = __builtin_amdgcn_wmma_f32_16x16x32_f16(
        /*neg_a=*/false, a.v, /*neg_b=*/false, b.v,
        /*c_mod=*/(short)0, acc, /*reuse_a=*/false, /*reuse_b=*/false);
  }

  int col = tn * 16 + r;
  if (col < Ncols) {
    int mbase = tm * 16 + g * 8;                // lanes 16-31 hold M rows +8
#pragma unroll
    for (int v = 0; v < 8; ++v) {
      int row = mbase + v;
      if (row < M) C[(size_t)row * ldc + col] = acc[v];
    }
  }
}

// ---------------- attention logits: al = <xp[n,h,:], a[h,:]> ----------------
__global__ void att_logits(const float* __restrict__ xp, const float* __restrict__ asrc,
                           const float* __restrict__ adst, float* __restrict__ als,
                           float* __restrict__ ald, int n, int H, int C, int ld) {
  int i = blockIdx.x * blockDim.x + threadIdx.x;
  if (i >= n * H) return;
  int node = i / H, h = i - node * H;
  const float* xr = xp + (size_t)node * ld + h * C;
  float s1 = 0.f, s2 = 0.f;
  for (int c = 0; c < C; ++c) {
    float v = xr[c];
    s1 += v * asrc[h * C + c];
    s2 += v * adst[h * C + c];
  }
  als[i] = s1;
  ald[i] = s2;
}

// ---------------- edge passes ----------------------------------------------
__device__ __forceinline__ void edge_sd(const int* __restrict__ ei, int e,
                                        int E, int& s, int& d) {
  if (e < E) { s = ei[e]; d = ei[E + e]; }
  else       { s = e - E; d = e - E; }           // self loops appended
}

__global__ void edge_max(const int* __restrict__ ei, const float* __restrict__ als,
                         const float* __restrict__ ald, unsigned* __restrict__ m,
                         int E, int ET, int H) {
  int i = blockIdx.x * blockDim.x + threadIdx.x;
  if (i >= ET * H) return;
  int e = i / H, h = i - e * H;
  int s, d; edge_sd(ei, e, E, s, d);
  float t = als[s * H + h] + ald[d * H + h];
  t = t > 0.f ? t : 0.2f * t;                    // leaky_relu, slope 0.2
  atomicMax(m + d * H + h, enc_f(t));
}

__global__ void edge_expsum(const int* __restrict__ ei, const float* __restrict__ als,
                            const float* __restrict__ ald, const unsigned* __restrict__ m,
                            float* __restrict__ sden, float* __restrict__ ee,
                            int E, int ET, int H) {
  int i = blockIdx.x * blockDim.x + threadIdx.x;
  if (i >= ET * H) return;
  int e = i / H, h = i - e * H;
  int s, d; edge_sd(ei, e, E, s, d);
  float t = als[s * H + h] + ald[d * H + h];
  t = t > 0.f ? t : 0.2f * t;
  float v = expf(t - dec_f(m[d * H + h]));
  ee[i] = v;
  atomicAdd(sden + d * H + h, v);
}

// out[dst, f] += ee[e, h(f)] * xp[src, f]   (unnormalized; divide in finalize)
__global__ void aggregate(const int* __restrict__ ei, const float* __restrict__ ee,
                          const float* __restrict__ xp, float* __restrict__ out,
                          int E, int ET, int H, int C, int HC, int ld) {
  int i = blockIdx.x * blockDim.x + threadIdx.x;
  if (i >= ET * HC) return;                      // 84.48M max, fits int32
  int e = i / HC, f = i - e * HC;
  int s, d; edge_sd(ei, e, E, s, d);
  int h = f / C;
  float w = ee[(size_t)e * H + h];
  atomicAdd(out + (size_t)d * HC + f, w * xp[(size_t)s * ld + f]);
}

// layers 1-2: normalize, +bias, ELU, cast to f16 A-matrix for next GEMM
__global__ void finalize12(const float* __restrict__ acc, const float* __restrict__ sden,
                           const float* __restrict__ bias, f16* __restrict__ xh,
                           int Nn, int NPad, int H, int C, int HC) {
  int i = blockIdx.x * blockDim.x + threadIdx.x;
  if (i >= NPad * HC) return;
  int node = i / HC, f = i - node * HC;
  if (node >= Nn) { xh[i] = (f16)0.0f; return; }
  int h = f / C;
  float v = acc[i] / (sden[node * H + h] + 1e-16f) + bias[f];
  v = v > 0.f ? v : (expf(v) - 1.f);             // ELU
  xh[i] = (f16)v;
}

// layer 3: normalize, +bias, log_softmax over 40 classes
__global__ void final3(const float* __restrict__ acc, const float* __restrict__ sden,
                       const float* __restrict__ bias, float* __restrict__ out, int Nn) {
  int n = blockIdx.x * blockDim.x + threadIdx.x;
  if (n >= Nn) return;
  float inv = 1.f / (sden[n] + 1e-16f);
  float vals[GOUT];
  float mx = -3.4e38f;
  for (int f = 0; f < GOUT; ++f) {
    float v = acc[(size_t)n * GOUT + f] * inv + bias[f];
    vals[f] = v;
    mx = v > mx ? v : mx;
  }
  float sum = 0.f;
  for (int f = 0; f < GOUT; ++f) sum += expf(vals[f] - mx);
  float lse = mx + logf(sum);
  for (int f = 0; f < GOUT; ++f) out[(size_t)n * GOUT + f] = vals[f] - lse;
}

// ---------------- host orchestration ---------------------------------------
static inline unsigned nb(long long n, int b) { return (unsigned)((n + b - 1) / b); }

extern "C" void kernel_launch(void* const* d_in, const int* in_sizes, int n_in,
                              void* d_out, int out_size, void* d_ws, size_t ws_size,
                              hipStream_t stream) {
  (void)in_sizes; (void)n_in; (void)out_size; (void)ws_size;
  const float* x   = (const float*)d_in[0];
  const int*   ei  = (const int*)d_in[1];
  const float* W1  = (const float*)d_in[2];
  const float* as1 = (const float*)d_in[3];
  const float* ad1 = (const float*)d_in[4];
  const float* b1  = (const float*)d_in[5];
  const float* W2  = (const float*)d_in[6];
  const float* as2 = (const float*)d_in[7];
  const float* ad2 = (const float*)d_in[8];
  const float* b2  = (const float*)d_in[9];
  const float* W3  = (const float*)d_in[10];
  const float* as3 = (const float*)d_in[11];
  const float* ad3 = (const float*)d_in[12];
  const float* b3  = (const float*)d_in[13];
  float* out = (float*)d_out;

  // static workspace carve (all 16B aligned), ~37.6 MB total
  char* ws = (char*)d_ws;
  f16*      xh   = (f16*)     (ws + 0);          // [GNPAD][256] f16
  f16*      wt   = (f16*)     (ws + 5128192);    // [256][256] f16 max
  float*    xp   = (float*)   (ws + 5259264);    // [GNPAD][256] f32
  float*    als  = (float*)   (ws + 15515648);   // [GN][8]
  float*    ald  = (float*)   (ws + 15835648);   // [GN][8]
  unsigned* mseg = (unsigned*)(ws + 16155648);   // [GN][8] encoded max
  float*    sden = (float*)   (ws + 16475648);   // [GN][8]
  float*    ee   = (float*)   (ws + 16795648);   // [GET][8]
  float*    hout = (float*)   (ws + 27355648);   // [GN][256]

  const int Mtiles = GNPAD / 16;                 // 626

  // ---------------- layer 1: IN=128 -> 8x32 concat, ELU ----------------
  cvt_pad_f16<<<nb((long long)GNPAD * GIN, 256), 256, 0, stream>>>(x, xh, GN, GNPAD, GIN);
  transpose_cvt_w<<<nb((long long)GHC * GIN, 256), 256, 0, stream>>>(W1, wt, GIN, GHC, GHC);
  {
    int tiles = Mtiles * (GHC / 16);
    gat_wmma_gemm<<<nb(tiles * 32, 128), 128, 0, stream>>>(xh, wt, xp, GN, GHC, GHC, GIN,
                                                           Mtiles, GHC / 16);
  }
  att_logits<<<nb((long long)GN * GH, 256), 256, 0, stream>>>(xp, as1, ad1, als, ald, GN, GH, GC, GHC);
  fill_f32<<<nb(GN * GH, 256), 256, 0, stream>>>((float*)mseg, GN * GH, 0.0f); // enc(-inf) < 0u impossible; 0u is identity
  fill_f32<<<nb(GN * GH, 256), 256, 0, stream>>>(sden, GN * GH, 0.0f);
  fill_f32<<<nb((long long)GN * GHC, 256), 256, 0, stream>>>(hout, GN * GHC, 0.0f);
  edge_max<<<nb((long long)GET * GH, 256), 256, 0, stream>>>(ei, als, ald, mseg, GE, GET, GH);
  edge_expsum<<<nb((long long)GET * GH, 256), 256, 0, stream>>>(ei, als, ald, mseg, sden, ee, GE, GET, GH);
  aggregate<<<nb((long long)GET * GHC, 256), 256, 0, stream>>>(ei, ee, xp, hout, GE, GET, GH, GC, GHC, GHC);
  finalize12<<<nb((long long)GNPAD * GHC, 256), 256, 0, stream>>>(hout, sden, b1, xh, GN, GNPAD, GH, GC, GHC);

  // ---------------- layer 2: 256 -> 8x32 concat, ELU ----------------
  transpose_cvt_w<<<nb((long long)GHC * GHC, 256), 256, 0, stream>>>(W2, wt, GHC, GHC, GHC);
  {
    int tiles = Mtiles * (GHC / 16);
    gat_wmma_gemm<<<nb(tiles * 32, 128), 128, 0, stream>>>(xh, wt, xp, GN, GHC, GHC, GHC,
                                                           Mtiles, GHC / 16);
  }
  att_logits<<<nb((long long)GN * GH, 256), 256, 0, stream>>>(xp, as2, ad2, als, ald, GN, GH, GC, GHC);
  fill_f32<<<nb(GN * GH, 256), 256, 0, stream>>>((float*)mseg, GN * GH, 0.0f);
  fill_f32<<<nb(GN * GH, 256), 256, 0, stream>>>(sden, GN * GH, 0.0f);
  fill_f32<<<nb((long long)GN * GHC, 256), 256, 0, stream>>>(hout, GN * GHC, 0.0f);
  edge_max<<<nb((long long)GET * GH, 256), 256, 0, stream>>>(ei, als, ald, mseg, GE, GET, GH);
  edge_expsum<<<nb((long long)GET * GH, 256), 256, 0, stream>>>(ei, als, ald, mseg, sden, ee, GE, GET, GH);
  aggregate<<<nb((long long)GET * GHC, 256), 256, 0, stream>>>(ei, ee, xp, hout, GE, GET, GH, GC, GHC, GHC);
  finalize12<<<nb((long long)GNPAD * GHC, 256), 256, 0, stream>>>(hout, sden, b2, xh, GN, GNPAD, GH, GC, GHC);

  // ---------------- layer 3: 256 -> 40, single head, log_softmax ----------------
  transpose_cvt_w<<<nb((long long)GOUTP * GHC, 256), 256, 0, stream>>>(W3, wt, GHC, GOUT, GOUTP);
  {
    int tiles = Mtiles * (GOUTP / 16);
    gat_wmma_gemm<<<nb(tiles * 32, 128), 128, 0, stream>>>(xh, wt, xp, GN, GOUT, GOUT, GHC,
                                                           Mtiles, GOUTP / 16);
  }
  att_logits<<<nb(GN, 256), 256, 0, stream>>>(xp, as3, ad3, als, ald, GN, 1, GOUT, GOUT);
  fill_f32<<<nb(GN, 256), 256, 0, stream>>>((float*)mseg, GN, 0.0f);
  fill_f32<<<nb(GN, 256), 256, 0, stream>>>(sden, GN, 0.0f);
  fill_f32<<<nb((long long)GN * GOUT, 256), 256, 0, stream>>>(hout, GN * GOUT, 0.0f);
  edge_max<<<nb(GET, 256), 256, 0, stream>>>(ei, als, ald, mseg, GE, GET, 1);
  edge_expsum<<<nb(GET, 256), 256, 0, stream>>>(ei, als, ald, mseg, sden, ee, GE, GET, 1);
  aggregate<<<nb((long long)GET * GOUT, 256), 256, 0, stream>>>(ei, ee, xp, hout, GE, GET, 1, GOUT, GOUT, GOUT);
  final3<<<nb(GN, 128), 128, 0, stream>>>(hout, sden, b3, out, GN);
}
